// TriangleMultiplicativeUpdate_20186346291619
// MI455X (gfx1250) — matrix operand order repaired and checked by reading, not compiled, not run on hardware
//
#include <hip/hip_runtime.h>
#include <cmath>

// ---------------------------------------------------------------------------
// TriangleMultiplicativeUpdate (outgoing) for MI455X / gfx1250 (wave32, WMMA)
// B=1, N=512, C_Z=C_H=128.
// All GEMMs run on v_wmma_f32_16x16x32_bf16 (bf16 operands, f32 accum).
// ---------------------------------------------------------------------------

static constexpr int kN = 512;
static constexpr int kC = 128;
static constexpr int kR = kN * kN;     // 262144 rows (flattened i,j)

typedef __attribute__((ext_vector_type(16))) __bf16 v16bf;
typedef __attribute__((ext_vector_type(8)))  __bf16 v8bf;
typedef __attribute__((ext_vector_type(4)))  __bf16 v4bf;
typedef __attribute__((ext_vector_type(8)))  float  v8f;
typedef __attribute__((ext_vector_type(4)))  float  v4f;

static __device__ inline v8f wmma_bf16(v16bf a, v16bf b, v8f c) {
  // (neg_a, A, neg_b, B, c_mod, C, reuse_a, reuse_b)
  return __builtin_amdgcn_wmma_f32_16x16x32_bf16(false, a, false, b, (short)0, c,
                                                 false, false);
}
static __device__ inline v8f vzero8() {
  v8f v = {0.f, 0.f, 0.f, 0.f, 0.f, 0.f, 0.f, 0.f};
  return v;
}
static __device__ inline float sigmoidf_(float x) { return 1.0f / (1.0f + expf(-x)); }

static __device__ inline v16bf pack2(v8bf lo, v8bf hi) {
  union { v16bf v; v8bf h[2]; } u;
  u.h[0] = lo; u.h[1] = hi;
  return u.v;
}

// A-fragment (16x32 bf16, MxK): lane m=lane&15; lanes<16 hold K 0-7 & 16-23,
// lanes>=16 hold K 8-15 & 24-31 -> two 16B chunks at k0+hi*8 and +16.
static __device__ inline v16bf load_afrag(const __bf16* base, int rowstride,
                                          int row, int k0, int hi) {
  const __bf16* p = base + (size_t)row * rowstride + k0 + hi * 8;
  return pack2(*(const v8bf*)p, *(const v8bf*)(p + 16));
}

// B-fragment (32x16 bf16, KxN) from weights stored [out][in] (contiguous K):
// lane n=lane&15; lanes<16 hold K 0-7 & 16-23, lanes>=16 hold K 8-15 & 24-31.
static __device__ inline v16bf load_wfrag(const __bf16* wt, int nt, int ks,
                                          int lm, int hi) {
  const __bf16* p = wt + (nt * 16 + lm) * kC + ks * 32 + hi * 8;
  return pack2(*(const v8bf*)p, *(const v8bf*)(p + 16));
}

// ---------------------------------------------------------------------------
// K0: weight prep  W_f32[in][out] -> Wt_bf16[out][in]   (6 matrices)
// ---------------------------------------------------------------------------
__global__ void __launch_bounds__(256)
k_prep_weights(const float* w0, const float* w1, const float* w2,
               const float* w3, const float* w4, const float* w5, __bf16* wt) {
  int idx = blockIdx.x * 256 + threadIdx.x;          // < 6*16384
  const float* ws[6] = {w0, w1, w2, w3, w4, w5};
  int m = idx >> 14;
  int rem = idx & 16383;
  int o = rem >> 7;
  int k = rem & 127;
  wt[idx] = (__bf16)ws[m][k * kC + o];
}

// ---------------------------------------------------------------------------
// K1: input LayerNorm, wave-per-row, zn bf16 [r][c]
// ---------------------------------------------------------------------------
__global__ void __launch_bounds__(256)
k_ln_in(const float* __restrict__ z, const float* __restrict__ gam,
        const float* __restrict__ bet, __bf16* __restrict__ zn) {
  int lane = threadIdx.x & 31, wave = threadIdx.x >> 5;
  int row = blockIdx.x * 8 + wave;
  const float* zr = z + (size_t)row * kC;
  v4f v = *(const v4f*)(zr + lane * 4);
  float s = v[0] + v[1] + v[2] + v[3];
  float ss = v[0] * v[0] + v[1] * v[1] + v[2] * v[2] + v[3] * v[3];
  for (int o = 16; o >= 1; o >>= 1) {
    s += __shfl_xor(s, o, 32);
    ss += __shfl_xor(ss, o, 32);
  }
  float mean = s * (1.f / kC);
  float rs = rsqrtf(ss * (1.f / kC) - mean * mean + 1e-5f);
  v4bf ov;
#pragma unroll
  for (int i2 = 0; i2 < 4; ++i2) {
    int c = lane * 4 + i2;
    ov[i2] = (__bf16)((v[i2] - mean) * rs * gam[c] + bet[c]);
  }
  *(v4bf*)(zn + (size_t)row * kC + lane * 4) = ov;
}

// ---------------------------------------------------------------------------
// K2: fused projections. One wave = 16 rows x all 128 cols x 5 weights.
// Columns processed in two halves of 4 tiles to keep <=8 live accumulators
// (64 VGPRs) -> no scratch spills inside the WMMA loop.
// a,b written channel-major [c][r] bf16 (LDS transpose); gate bf16 [r][c].
// ---------------------------------------------------------------------------
__global__ void __launch_bounds__(256)
k_proj(const __bf16* __restrict__ zn, const float* __restrict__ mask,
       const __bf16* __restrict__ wt,
       const float* __restrict__ b_ag, const float* __restrict__ b_ap,
       const float* __restrict__ b_bg, const float* __restrict__ b_bp,
       const float* __restrict__ b_g,
       __bf16* __restrict__ a_out, __bf16* __restrict__ b_out,
       __bf16* __restrict__ gate_out) {
  __shared__ __align__(32) __bf16 lt[8][kC][16];     // 32 KB transpose staging
  int lane = threadIdx.x & 31, wave = threadIdx.x >> 5;
  int lm = lane & 15, hi = lane >> 4;
  int rbase = (blockIdx.x * 8 + wave) * 16;

  v16bf afr[4];
#pragma unroll
  for (int ks = 0; ks < 4; ++ks) afr[ks] = load_afrag(zn, kC, rbase + lm, ks * 32, hi);
  float mrow[8];
#pragma unroll
  for (int g = 0; g < 8; ++g) mrow[g] = mask[rbase + g + hi * 8];

  const float* bg_arr[2] = {b_ag, b_bg};
  const float* bp_arr[2] = {b_ap, b_bp};
  __bf16* out_arr[2] = {a_out, b_out};

  for (int pass = 0; pass < 2; ++pass) {
    const __bf16* wtg = wt + (pass * 2 + 0) * (kC * kC);
    const __bf16* wtp = wt + (pass * 2 + 1) * (kC * kC);
    for (int half = 0; half < 2; ++half) {
      v8f accg[4], accp[4];
#pragma unroll
      for (int q = 0; q < 4; ++q) { accg[q] = vzero8(); accp[q] = vzero8(); }
#pragma unroll
      for (int ks = 0; ks < 4; ++ks) {
#pragma unroll
        for (int q = 0; q < 4; ++q) {
          int nt = half * 4 + q;
          accg[q] = wmma_bf16(afr[ks], load_wfrag(wtg, nt, ks, lm, hi), accg[q]);
          accp[q] = wmma_bf16(afr[ks], load_wfrag(wtp, nt, ks, lm, hi), accp[q]);
        }
      }
#pragma unroll
      for (int q = 0; q < 4; ++q) {
        int nt = half * 4 + q;
        float big = bg_arr[pass][nt * 16 + lm];
        float bip = bp_arr[pass][nt * 16 + lm];
#pragma unroll
        for (int g = 0; g < 8; ++g) {
          float val = mrow[g] * sigmoidf_(accg[q][g] + big) * (accp[q][g] + bip);
          lt[wave][nt * 16 + lm][g + hi * 8] = (__bf16)val;
        }
      }
    }
    __syncthreads();
    __bf16* op = out_arr[pass];
#pragma unroll
    for (int q = 0; q < 4; ++q) {
      int c = q * 32 + lane;
      v16bf t = *(const v16bf*)&lt[wave][c][0];
      *(v16bf*)(op + (size_t)c * kR + rbase) = t;   // channel-major, 32B/lane
    }
    __syncthreads();
  }

  {  // gate = sigmoid(zn @ w_g + b_g), natural [r][c] layout, bf16
    const __bf16* wtg = wt + 4 * (kC * kC);
    for (int half = 0; half < 2; ++half) {
      v8f acc[4];
#pragma unroll
      for (int q = 0; q < 4; ++q) acc[q] = vzero8();
#pragma unroll
      for (int ks = 0; ks < 4; ++ks)
#pragma unroll
        for (int q = 0; q < 4; ++q)
          acc[q] = wmma_bf16(afr[ks], load_wfrag(wtg, half * 4 + q, ks, lm, hi), acc[q]);
#pragma unroll
      for (int q = 0; q < 4; ++q) {
        int nt = half * 4 + q;
        float bi = b_g[nt * 16 + lm];
#pragma unroll
        for (int g = 0; g < 8; ++g) {
          float val = sigmoidf_(acc[q][g] + bi);
          gate_out[(size_t)(rbase + g + hi * 8) * kC + nt * 16 + lm] = (__bf16)val;
        }
      }
    }
  }
}

// ---------------------------------------------------------------------------
// K3: triangle einsum. Per block (256 thr / 8 waves): one channel, 128x128
// (i,j) macro-tile, K=512 in 8 stages of 64. Software-pipelined: stage s+1
// global loads issue before stage s compute; 16 WMMAs/wave between barriers.
// a/b are channel-major -> contiguous K rows; a+b (134MB bf16) sit in L2.
// ---------------------------------------------------------------------------
__global__ void __launch_bounds__(256)
k_einsum(const __bf16* __restrict__ a_ws, const __bf16* __restrict__ b_ws,
         float* __restrict__ x_ws) {
  __shared__ __align__(32) __bf16 Asub[128][64];   // 16 KB
  __shared__ __align__(32) __bf16 Bsub[128][64];   // 16 KB
  int t = threadIdx.x;                   // 256 threads = 8 waves
  int lane = t & 31, wave = t >> 5;
  int lm = lane & 15, hi = lane >> 4;
  int bx = blockIdx.x;
  int ch = bx >> 4;
  int it = (bx >> 2) & 3, jt = bx & 3;
  int ibase = it * 128, jbase = jt * 128;
  const __bf16* ab = a_ws + (size_t)ch * kR;
  const __bf16* bb = b_ws + (size_t)ch * kR;
  int lrow = t >> 1;                    // 0..127: tile row loaded by this thread
  int lseg = (t & 1) * 32;              // element offset within row (0 or 32)

  v8f acc[8];
#pragma unroll
  for (int nt = 0; nt < 8; ++nt) acc[nt] = vzero8();

  const __bf16* agp = ab + (size_t)(ibase + lrow) * kN + lseg;
  const __bf16* bgp = bb + (size_t)(jbase + lrow) * kN + lseg;

  // preload stage 0
  v16bf ra[2], rb[2];
#pragma unroll
  for (int q = 0; q < 2; ++q) {
    ra[q] = *(const v16bf*)(agp + q * 16);
    rb[q] = *(const v16bf*)(bgp + q * 16);
  }

  for (int s = 0; s < 8; ++s) {
    __syncthreads();                     // previous compute done, safe to overwrite
#pragma unroll
    for (int q = 0; q < 2; ++q) {
      *(v16bf*)&Asub[lrow][lseg + q * 16] = ra[q];
      *(v16bf*)&Bsub[lrow][lseg + q * 16] = rb[q];
    }
    __syncthreads();                     // tile visible to all waves

    if (s < 7) {                         // issue next-stage loads before compute
      int k0 = (s + 1) * 64;
#pragma unroll
      for (int q = 0; q < 2; ++q) {
        ra[q] = *(const v16bf*)(agp + k0 + q * 16);
        rb[q] = *(const v16bf*)(bgp + k0 + q * 16);
      }
      if (s < 6) {                       // prefetch one stage further ahead
        __builtin_prefetch(agp + k0 + 64, 0, 0);
        __builtin_prefetch(bgp + k0 + 64, 0, 0);
      }
    }

#pragma unroll
    for (int kk = 0; kk < 2; ++kk) {
      const __bf16* ar = &Asub[wave * 16 + lm][kk * 32 + hi * 8];
      v16bf af = pack2(*(const v8bf*)ar, *(const v8bf*)(ar + 16));
#pragma unroll
      for (int nt = 0; nt < 8; ++nt) {
        const __bf16* br = &Bsub[nt * 16 + lm][kk * 32 + hi * 8];
        v16bf bf_ = pack2(*(const v8bf*)br, *(const v8bf*)(br + 16));
        acc[nt] = wmma_bf16(af, bf_, acc[nt]);
      }
    }
  }

#pragma unroll
  for (int nt = 0; nt < 8; ++nt)
#pragma unroll
    for (int g = 0; g < 8; ++g)
      x_ws[(size_t)ch * kR + (size_t)(ibase + wave * 16 + g + hi * 8) * kN +
           jbase + nt * 16 + lm] = acc[nt][g];
}

// ---------------------------------------------------------------------------
// K4: gather x over c, LN, @ w_z, * gate, write f32 output [i,j,c]
// ---------------------------------------------------------------------------
__global__ void __launch_bounds__(128)
k_out(const float* __restrict__ x_ws, const __bf16* __restrict__ wt_z,
      const float* __restrict__ b_z, const float* __restrict__ lng,
      const float* __restrict__ lnb, const __bf16* __restrict__ gate,
      float* __restrict__ out) {
  __shared__ float xt[4][kC][16];                 // 32 KB
  __shared__ __align__(32) __bf16 znt[4][16][kC]; // 16 KB
  int lane = threadIdx.x & 31, wave = threadIdx.x >> 5;
  int lm = lane & 15, hi = lane >> 4;
  int rbase = (blockIdx.x * 4 + wave) * 16;       // 16 rows, same i, j0..j0+15
  int i = rbase >> 9, j0 = rbase & 511;

#pragma unroll
  for (int q = 0; q < 4; ++q) {
    int c = q * 32 + lane;
    const float* src = x_ws + (size_t)c * kR + (size_t)i * kN + j0;
#pragma unroll
    for (int jj = 0; jj < 4; ++jj)
      *(v4f*)&xt[wave][c][jj * 4] = *(const v4f*)(src + jj * 4);
  }
  __syncthreads();

  float s = 0.f, ss = 0.f;
  for (int c = hi * 64; c < hi * 64 + 64; ++c) {
    float v = xt[wave][c][lm];
    s += v; ss += v * v;
  }
  s += __shfl_xor(s, 16, 32);
  ss += __shfl_xor(ss, 16, 32);
  float mean = s * (1.f / kC);
  float rs = rsqrtf(ss * (1.f / kC) - mean * mean + 1e-5f);
  for (int c = hi * 64; c < hi * 64 + 64; ++c)
    znt[wave][lm][c] = (__bf16)((xt[wave][c][lm] - mean) * rs * lng[c] + lnb[c]);
  __syncthreads();

  v8f acc[8];
#pragma unroll
  for (int nt = 0; nt < 8; ++nt) acc[nt] = vzero8();
#pragma unroll
  for (int ks = 0; ks < 4; ++ks) {
    const __bf16* ar = &znt[wave][lm][ks * 32 + hi * 8];
    v16bf af = pack2(*(const v8bf*)ar, *(const v8bf*)(ar + 16));
#pragma unroll
    for (int nt = 0; nt < 8; ++nt)
      acc[nt] = wmma_bf16(af, load_wfrag(wt_z, nt, ks, lm, hi), acc[nt]);
  }
#pragma unroll
  for (int nt = 0; nt < 8; ++nt) {
    float bz = b_z[nt * 16 + lm];
#pragma unroll
    for (int g = 0; g < 8; ++g) {
      int r = rbase + g + hi * 8;
      int co = nt * 16 + lm;
      out[(size_t)r * kC + co] = (acc[nt][g] + bz) * (float)gate[(size_t)r * kC + co];
    }
  }
}

// ---------------------------------------------------------------------------
extern "C" void kernel_launch(void* const* d_in, const int* in_sizes, int n_in,
                              void* d_out, int out_size, void* d_ws, size_t ws_size,
                              hipStream_t stream) {
  const float* z    = (const float*)d_in[0];
  const float* mask = (const float*)d_in[1];
  const float* w_ag = (const float*)d_in[2];
  const float* b_ag = (const float*)d_in[3];
  const float* w_ap = (const float*)d_in[4];
  const float* b_ap = (const float*)d_in[5];
  const float* w_bg = (const float*)d_in[6];
  const float* b_bg = (const float*)d_in[7];
  const float* w_bp = (const float*)d_in[8];
  const float* b_bp = (const float*)d_in[9];
  const float* w_g  = (const float*)d_in[10];
  const float* b_g  = (const float*)d_in[11];
  const float* w_z  = (const float*)d_in[12];
  const float* b_z  = (const float*)d_in[13];
  const float* lnig = (const float*)d_in[14];
  const float* lnib = (const float*)d_in[15];
  const float* lnog = (const float*)d_in[16];
  const float* lnob = (const float*)d_in[17];

  // Workspace layout (bytes):
  //   wt   : 6*128*128*2      =    196,608
  //   zn   : R*C bf16         = 67,108,864
  //   a_ws : C*R bf16         = 67,108,864   (channel-major)
  //   b_ws : C*R bf16         = 67,108,864   (channel-major)
  //   g_ws : R*C bf16         = 67,108,864
  //   x_ws : C*R f32          =134,217,728   (channel-major)
  // total ~403 MB
  char* ws = (char*)d_ws;
  const size_t SB = 67108864ull;
  __bf16* wt   = (__bf16*)ws;
  __bf16* zn   = (__bf16*)(ws + 196608);
  __bf16* a_ws = (__bf16*)(ws + 196608 + 1 * SB);
  __bf16* b_ws = (__bf16*)(ws + 196608 + 2 * SB);
  __bf16* g_ws = (__bf16*)(ws + 196608 + 3 * SB);
  float*  x_ws = (float*)(ws + 196608 + 4 * SB);
  float* out = (float*)d_out;

  k_prep_weights<<<384, 256, 0, stream>>>(w_ag, w_ap, w_bg, w_bp, w_g, w_z, wt);
  k_ln_in<<<kR / 8, 256, 0, stream>>>(z, lnig, lnib, zn);
  k_proj<<<kR / 16 / 8, 256, 0, stream>>>(zn, mask, wt, b_ag, b_ap, b_bg, b_bp,
                                          b_g, a_ws, b_ws, g_ws);
  k_einsum<<<kC * 16, 256, 0, stream>>>(a_ws, b_ws, x_ws);
  k_out<<<kR / 16 / 4, 128, 0, stream>>>(x_ws, wt + 5 * kC * kC, b_z, lnog, lnob,
                                         g_ws, out);
}
